// sLSTMLayer_47777216201153
// MI455X (gfx1250) — compile-verified, matrix operand order checked by reading, not run on hardware
//
#include <hip/hip_runtime.h>
#include <hip/hip_bf16.h>
#include <math.h>

typedef __bf16 bf16_t;
typedef __attribute__((ext_vector_type(16))) __bf16 v16bf;
typedef __attribute__((ext_vector_type(8)))  __bf16 v8bf;
typedef __attribute__((ext_vector_type(8)))  float  v8f;

#define B_  4096
#define F_  1024
#define H_  4
#define D_  256
#define K_  4
#define BF_ (B_ * F_)
#define MAT_ (H_ * D_ * D_)   // 262144 elements per weight matrix

// ---------------------------------------------------------------------------
// Kernel 1: conv shift + SiLU + bf16 packing of conv_act / inputs / h
// ---------------------------------------------------------------------------
__global__ __launch_bounds__(256) void conv_pack_kernel(
    const float* __restrict__ inputs, const float* __restrict__ h,
    const float* __restrict__ conv_state, const float* __restrict__ ck,
    const float* __restrict__ cb,
    float* __restrict__ ncs_out,          // new_conv_state (B,K,F) in d_out
    bf16_t* __restrict__ conv_bf, bf16_t* __restrict__ in_bf,
    bf16_t* __restrict__ h_bf)
{
    int idx = blockIdx.x * 256 + threadIdx.x;
    if (idx >= BF_) return;
    int b = idx >> 10;
    int f = idx & (F_ - 1);

    const float* cs = conv_state + (size_t)b * (K_ * F_);
    float s1 = cs[1 * F_ + f];
    float s2 = cs[2 * F_ + f];
    float s3 = cs[3 * F_ + f];
    float x  = inputs[idx];

    float* ncs = ncs_out + (size_t)b * (K_ * F_);
    ncs[0 * F_ + f] = s1;
    ncs[1 * F_ + f] = s2;
    ncs[2 * F_ + f] = s3;
    ncs[3 * F_ + f] = x;

    float cx = s1 * ck[0 * F_ + f] + s2 * ck[1 * F_ + f] +
               s3 * ck[2 * F_ + f] + x  * ck[3 * F_ + f] + cb[f];
    float act = cx / (1.0f + __expf(-cx));   // SiLU

    conv_bf[idx] = (bf16_t)act;
    in_bf[idx]   = (bf16_t)x;
    h_bf[idx]    = (bf16_t)h[idx];
}

// ---------------------------------------------------------------------------
// Kernel 2: weight convert + transpose to bf16, N-major (Wt[g][h][e][d])
// ---------------------------------------------------------------------------
__global__ __launch_bounds__(256) void wconv_kernel(
    const float* __restrict__ w0, const float* __restrict__ w1,
    const float* __restrict__ w2, const float* __restrict__ w3,
    const float* __restrict__ w4, const float* __restrict__ w5,
    const float* __restrict__ w6, const float* __restrict__ w7,
    bf16_t* __restrict__ wt)
{
    int idx = blockIdx.x * 256 + threadIdx.x;
    if (idx >= 8 * MAT_) return;
    const float* wsrc[8] = {w0, w1, w2, w3, w4, w5, w6, w7};
    int mi = idx >> 18;            // which of 8 matrices (MAT_ = 2^18)
    int r  = idx & (MAT_ - 1);     // flat (h,d,e)
    int hh = r >> 16;
    int dd = (r >> 8) & 255;
    int ee = r & 255;
    float v = wsrc[mi][r];
    wt[(size_t)mi * MAT_ + (size_t)hh * (D_ * D_) + (size_t)ee * D_ + dd] = (bf16_t)v;
}

// ---------------------------------------------------------------------------
// WMMA fragment loader: 16-bit A-layout (ISA 7.12.2).
// lane 0-15  : row = row0+lane,     elems 0..7 -> K k0..k0+7,  8..15 -> K k0+16..23
// lane 16-31 : row = row0+lane-16,  elems 0..7 -> K k0+8..15,  8..15 -> K k0+24..31
// B-fragments use the same mapping with N playing the role of M (weights are
// stored N-major so K is contiguous).
// ---------------------------------------------------------------------------
__device__ __forceinline__ v16bf load_frag(const bf16_t* __restrict__ base,
                                           int stride, int lane, int row0, int k0)
{
    int r  = row0 + (lane & 15);
    int kb = k0 + ((lane >> 4) << 3);
    const bf16_t* p = base + (size_t)r * stride + kb;
    v8bf lo = *(const v8bf*)p;
    v8bf hi = *(const v8bf*)(p + 16);
    return __builtin_shufflevector(lo, hi, 0, 1, 2, 3, 4, 5, 6, 7,
                                   8, 9, 10, 11, 12, 13, 14, 15);
}

// ---------------------------------------------------------------------------
// Kernel 3: block-diagonal GEMM per (head, gate):
//   pre_g[b, h*256+n] = sum_k X[b, h*256+k]*W[h][k][n] + sum_k hprev*R
// Workgroup: 128(M) x 128(N) tile; 8 waves in 4(M) x 2(N); wave = 32x64.
// ---------------------------------------------------------------------------
__global__ __launch_bounds__(256) void gemm_gate_kernel(
    const bf16_t* __restrict__ convA,   // B x F bf16 (SiLU(conv))
    const bf16_t* __restrict__ inA,     // B x F bf16 (inputs)
    const bf16_t* __restrict__ hA,      // B x F bf16 (h)
    const bf16_t* __restrict__ wt,      // 8 x H x 256 x 256 bf16, N-major
    float* __restrict__ preAll)         // 4 x B x F f32 pre-activations
{
    const int hz   = blockIdx.z;
    const int head = hz >> 2;
    const int gate = hz & 3;

    const bf16_t* X  = (gate < 2) ? convA : inA;       // i,f use conv_act; z,o use inputs
    const bf16_t* Wg = wt + (size_t)gate       * MAT_ + (size_t)head * (D_ * D_);
    const bf16_t* Rg = wt + (size_t)(4 + gate) * MAT_ + (size_t)head * (D_ * D_);
    float* pre = preAll + (size_t)gate * BF_;

    const int lane = threadIdx.x & 31;
    const int w    = threadIdx.x >> 5;
    const int wm   = w >> 1;                  // 0..3
    const int wn   = w & 1;                   // 0..1
    const int mwave  = blockIdx.x * 128 + wm * 32;      // batch-row base
    const int nlocal = blockIdx.y * 128 + wn * 64;      // column within head

    v8f acc[2][4] = {};

    #pragma unroll
    for (int phase = 0; phase < 2; ++phase) {
        const bf16_t* A  = phase ? hA : X;
        const bf16_t* Bm = phase ? Rg : Wg;
        const bf16_t* Abase = A + head * D_;            // head column offset

        for (int k0 = 0; k0 < D_; k0 += 32) {
            v16bf af[2], bfr[4];
            #pragma unroll
            for (int mt = 0; mt < 2; ++mt)
                af[mt] = load_frag(Abase, F_, lane, mwave + mt * 16, k0);
            #pragma unroll
            for (int nt = 0; nt < 4; ++nt)
                bfr[nt] = load_frag(Bm, D_, lane, nlocal + nt * 16, k0);

            #pragma unroll
            for (int mt = 0; mt < 2; ++mt)
                #pragma unroll
                for (int nt = 0; nt < 4; ++nt)
                    acc[mt][nt] = __builtin_amdgcn_wmma_f32_16x16x32_bf16(
                        false, af[mt], false, bfr[nt],
                        (short)0, acc[mt][nt], false, false);
        }
    }

    // C/D layout: VGPR r <-> M = r (lanes 0-15) / M = 8+r (lanes 16-31); N = lane&15
    const int nglob = head * D_ + nlocal;
    #pragma unroll
    for (int mt = 0; mt < 2; ++mt) {
        #pragma unroll
        for (int nt = 0; nt < 4; ++nt) {
            int n  = nglob + nt * 16 + (lane & 15);
            int m0 = mwave + mt * 16 + ((lane >> 4) << 3);
            #pragma unroll
            for (int r = 0; r < 8; ++r)
                pre[(size_t)(m0 + r) * F_ + n] = acc[mt][nt][r];
        }
    }
}

// ---------------------------------------------------------------------------
// Kernel 4: sLSTM pointwise + stabilizer + per-head LayerNorm.
// One 256-thread block per (b, head); thread d owns element f = head*256+d.
// ---------------------------------------------------------------------------
__global__ __launch_bounds__(256) void epilogue_kernel(
    const float* __restrict__ preAll,
    const float* __restrict__ c, const float* __restrict__ n,
    const float* __restrict__ m,
    const float* __restrict__ bi, const float* __restrict__ bfb,
    const float* __restrict__ bz, const float* __restrict__ bo,
    const float* __restrict__ lns,
    float* __restrict__ out)    // d_out: [out | c_new | n_new | m_new | h_new | ncs]
{
    __shared__ float s_sum[256];
    __shared__ float s_sq[256];

    const int b    = blockIdx.x;
    const int head = blockIdx.y;
    const int d    = threadIdx.x;
    const int f    = head * D_ + d;
    const size_t idx = (size_t)b * F_ + f;

    float iv = preAll[0 * (size_t)BF_ + idx] + bi[f];
    float fv = preAll[1 * (size_t)BF_ + idx] + bfb[f];
    float zv = preAll[2 * (size_t)BF_ + idx] + bz[f];
    float ov = preAll[3 * (size_t)BF_ + idx] + bo[f];

    float og = 1.0f / (1.0f + __expf(-ov));                       // sigmoid(o)
    float lf = fminf(fv, 0.0f) - log1pf(__expf(-fabsf(fv)));      // -softplus(-f)

    float mv = m[idx];
    float mn = fmaxf(lf + mv, iv);                                // general branch (n != 0)
    float ip = fminf(__expf(iv - mn), 1.0f);
    float fp = fminf(__expf(lf + mv - mn), 1.0f);

    float cn = fp * c[idx] + ip * tanhf(zv);
    float nn = fp * n[idx] + ip;
    float hn = og * (cn / fmaxf(nn, 1e-6f));

    // LayerNorm over head_dim (D=256)
    s_sum[d] = hn;
    s_sq[d]  = hn * hn;
    __syncthreads();
    #pragma unroll
    for (int s = 128; s > 0; s >>= 1) {
        if (d < s) {
            s_sum[d] += s_sum[d + s];
            s_sq[d]  += s_sq[d + s];
        }
        __syncthreads();
    }
    float mu  = s_sum[0] * (1.0f / D_);
    float var = s_sq[0] * (1.0f / D_) - mu * mu;
    float y   = (hn - mu) * rsqrtf(var + 1e-6f) * lns[f];

    out[0 * (size_t)BF_ + idx] = y;
    out[1 * (size_t)BF_ + idx] = cn;
    out[2 * (size_t)BF_ + idx] = nn;
    out[3 * (size_t)BF_ + idx] = mn;
    out[4 * (size_t)BF_ + idx] = hn;
}

// ---------------------------------------------------------------------------
extern "C" void kernel_launch(void* const* d_in, const int* in_sizes, int n_in,
                              void* d_out, int out_size, void* d_ws, size_t ws_size,
                              hipStream_t stream)
{
    const float* inputs      = (const float*)d_in[0];
    const float* c           = (const float*)d_in[1];
    // d_in[2] = n, d_in[3] = m
    const float* n           = (const float*)d_in[2];
    const float* m           = (const float*)d_in[3];
    const float* h           = (const float*)d_in[4];
    const float* conv_state  = (const float*)d_in[5];
    const float* conv_kernel = (const float*)d_in[6];
    const float* conv_bias   = (const float*)d_in[7];
    const float* Wi = (const float*)d_in[8];
    const float* Wf = (const float*)d_in[9];
    const float* Wz = (const float*)d_in[10];
    const float* Wo = (const float*)d_in[11];
    const float* Ri = (const float*)d_in[12];
    const float* Rf = (const float*)d_in[13];
    const float* Rz = (const float*)d_in[14];
    const float* Ro = (const float*)d_in[15];
    const float* bi = (const float*)d_in[16];
    const float* bfb = (const float*)d_in[17];
    const float* bz = (const float*)d_in[18];
    const float* bo = (const float*)d_in[19];
    const float* lns = (const float*)d_in[20];

    float* out = (float*)d_out;
    float* ncs_out = out + 5 * (size_t)BF_;   // new_conv_state region

    // Workspace layout (bytes):
    char* ws = (char*)d_ws;
    bf16_t* conv_bf = (bf16_t*)(ws);                            // BF * 2
    bf16_t* in_bf   = (bf16_t*)(ws + (size_t)BF_ * 2);          // BF * 2
    bf16_t* h_bf    = (bf16_t*)(ws + (size_t)BF_ * 4);          // BF * 2
    bf16_t* wt      = (bf16_t*)(ws + (size_t)BF_ * 6);          // 8*MAT_*2 = 4 MB
    float*  preAll  = (float*) (ws + (size_t)BF_ * 6 + (size_t)8 * MAT_ * 2); // 4*BF*4

    // 1) conv shift + SiLU + bf16 pack
    conv_pack_kernel<<<BF_ / 256, 256, 0, stream>>>(
        inputs, h, conv_state, conv_kernel, conv_bias,
        ncs_out, conv_bf, in_bf, h_bf);

    // 2) weight convert + transpose
    wconv_kernel<<<(8 * MAT_) / 256, 256, 0, stream>>>(
        Wi, Wf, Wz, Wo, Ri, Rf, Rz, Ro, wt);

    // 3) WMMA GEMMs: grid = (M tiles, N tiles per head, head*4+gate)
    gemm_gate_kernel<<<dim3(B_ / 128, D_ / 128, H_ * 4), 256, 0, stream>>>(
        conv_bf, in_bf, h_bf, wt, preAll);

    // 4) pointwise + layernorm epilogue
    epilogue_kernel<<<dim3(B_, H_), 256, 0, stream>>>(
        preAll, c, n, m, bi, bfb, bz, bo, lns, out);
}
